// InteractionNetwork_38809324487026
// MI455X (gfx1250) — matrix operand order repaired and checked by reading, not compiled, not run on hardware
//
#include <hip/hip_runtime.h>

// ---------------------------------------------------------------------------
// InteractionNetwork forward for MI455X (gfx1250, wave32, WMMA bf16)
//   edge MLP (288->128->128, relu)  -> scatter-add to agg
//   node block (256->128 gate+update) -> x_weights out, pooled scatter
//   global block (192->64) -> u_new out
// Edge kernel: each wave owns TWO 16-row A-tiles (M=32) so every B fragment
// load feeds two v_wmma ops; B window preloaded per k-step.
// ---------------------------------------------------------------------------

#define N_NODES  50000
#define N_EDGES  800000
#define N_GRAPHS 512
#define D_IN     128
#define D_EDGE   32
#define D_HID    128
#define D_GLOB   64

#define KE   (2 * D_IN + D_EDGE)   // 288  edge-MLP K
#define KN   (D_IN + D_HID)        // 256  node-block K
#define ROW1 (KE + 8)              // 296 halves = 592B (16B multiple)
#define ROW2 (D_HID + 8)           // 136 halves = 272B
#define ROW3 (KN + 8)              // 264 halves = 528B

typedef __bf16 bf16_t;
typedef __attribute__((ext_vector_type(16))) __bf16 v16bf;
typedef __attribute__((ext_vector_type(8)))  __bf16 v8bf;
typedef __attribute__((ext_vector_type(8)))  float  v8f;

// float -> bf16 (round to nearest even), no reliance on native casts
__device__ __forceinline__ unsigned short bfbits(float f) {
  unsigned u = __builtin_bit_cast(unsigned, f);
  unsigned r = u + 0x7FFFu + ((u >> 16) & 1u);
  return (unsigned short)(r >> 16);
}
__device__ __forceinline__ bf16_t f2bf(float f) {
  unsigned short h = bfbits(f);
  return __builtin_bit_cast(bf16_t, h);
}
__device__ __forceinline__ unsigned long long pack4(float4 v) {
  return (unsigned long long)bfbits(v.x) |
         ((unsigned long long)bfbits(v.y) << 16) |
         ((unsigned long long)bfbits(v.z) << 32) |
         ((unsigned long long)bfbits(v.w) << 48);
}

// A-fragment loader: 16x32 bf16 tile, ISA layout.
// lane<16: halves 0..7 = K 0..7, halves 8..15 = K 16..23 (of the 32-step)
// lane>=16: K 8..15 and K 24..31.  row = per-lane LDS row base (16B aligned).
__device__ __forceinline__ v16bf load_afrag(const bf16_t* __restrict__ row,
                                            int ks, int lane) {
  const int kb = (lane >> 4) * 8;
  v8bf a0 = *(const v8bf*)(row + ks * 32 + kb);       // ds_load_b128
  v8bf a1 = *(const v8bf*)(row + ks * 32 + kb + 16);  // ds_load_b128
  v16bf a;
#pragma unroll
  for (int i = 0; i < 8; ++i) { a[i] = a0[i]; a[i + 8] = a1[i]; }
  return a;
}

#define WMMA_BF16(A, B, C)                                                  \
  __builtin_amdgcn_wmma_f32_16x16x32_bf16(false, (A), false, (B), (short)0, \
                                          (C), false, false)

// ---------------------------------------------------------------------------
// Weight pre-swizzle: W is (ktiles*32) x 128 row-major f32.
// Output: fragment-major bf16 so a lane's 16 B-halves are one 32B load.
// frag[((ks*8+nt)*32+lane)*16 + i] = W[ks*32 + (lane>>4)*8 + (i<8 ? i : i+8)][nt*16 + (lane&15)]
// ---------------------------------------------------------------------------
__global__ void in455_prep_w(const float* __restrict__ W,
                             bf16_t* __restrict__ frag, int ktiles) {
  int tid = blockIdx.x * blockDim.x + threadIdx.x;
  int total = ktiles * 8 * 32;
  if (tid >= total) return;
  int lane = tid & 31, f = tid >> 5;
  int nt = f & 7, ks = f >> 3;
  int n  = nt * 16 + (lane & 15);
  int kb = ks * 32 + (lane >> 4) * 8;
#pragma unroll
  for (int i = 0; i < 16; ++i)
    frag[(size_t)tid * 16 + i] = f2bf(W[(size_t)(kb + (i < 8 ? i : i + 8)) * 128 + n]);
}

__global__ void in455_zero(float* __restrict__ p, size_t n) {
  size_t i = blockIdx.x * (size_t)blockDim.x + threadIdx.x;
  size_t s = gridDim.x * (size_t)blockDim.x;
  for (; i < n; i += s) p[i] = 0.0f;
}

// ---------------------------------------------------------------------------
// Edge block: per wave, a 32-edge tile (two 16-row WMMA tiles).  Two fused
// bf16 WMMA GEMM layers, then atomic scatter into agg[dest].
// ---------------------------------------------------------------------------
__global__ __launch_bounds__(64) void in455_edge(
    const float* __restrict__ x, const int* __restrict__ eidx,
    const float* __restrict__ eattr,
    const bf16_t* __restrict__ We1f, const float* __restrict__ be1,
    const bf16_t* __restrict__ We2f, const float* __restrict__ be2,
    float* __restrict__ agg) {
  __shared__ __align__(16) bf16_t sIn[2][32][ROW1];
  __shared__ __align__(16) bf16_t sHid[2][32][ROW2];
  __shared__ int sDst[2][32];

  const int lane = threadIdx.x & 31;
  const int wv   = threadIdx.x >> 5;
  const int mlo  = lane & 15;
  const int khi  = lane >> 4;
  const int tile = (blockIdx.x * 2 + wv) * 32;  // 800000/32 tiles exactly

  // ---- gather: concat(src, dest, edge_attr) -> bf16 in LDS --------------
  sDst[wv][lane] = eidx[N_EDGES + tile + lane];
#pragma unroll 4
  for (int m = 0; m < 32; ++m) {
    const int e = tile + m;
    const int s = eidx[e];
    const int d = eidx[N_EDGES + e];
    float4 vs = ((const float4*)(x + (size_t)s * D_IN))[lane];
    float4 vd = ((const float4*)(x + (size_t)d * D_IN))[lane];
    *(unsigned long long*)&sIn[wv][m][lane * 4]        = pack4(vs);
    *(unsigned long long*)&sIn[wv][m][D_IN + lane * 4] = pack4(vd);
    sIn[wv][m][2 * D_IN + lane] = f2bf(eattr[(size_t)e * D_EDGE + lane]);
  }
  __syncthreads();

  // ---- layer 1: (32x288) @ (288x128), K-steps = 9 ------------------------
  const bf16_t* arow0 = &sIn[wv][mlo][0];
  const bf16_t* arow1 = &sIn[wv][16 + mlo][0];
  const v16bf* W1 = (const v16bf*)We1f;
  v8f acc0[8] = {}, acc1[8] = {};
  for (int ks = 0; ks < 9; ++ks) {
    v16bf a0 = load_afrag(arow0, ks, lane);
    v16bf a1 = load_afrag(arow1, ks, lane);
    v16bf b[8];
#pragma unroll
    for (int nt = 0; nt < 8; ++nt) b[nt] = W1[(ks * 8 + nt) * 32 + lane];
#pragma unroll
    for (int nt = 0; nt < 8; ++nt) {
      acc0[nt] = WMMA_BF16(a0, b[nt], acc0[nt]);
      acc1[nt] = WMMA_BF16(a1, b[nt], acc1[nt]);
    }
  }
  // bias + relu, stage hidden (C/D layout -> row-major bf16 in LDS)
#pragma unroll
  for (int nt = 0; nt < 8; ++nt) {
    const float bias = be1[nt * 16 + mlo];
#pragma unroll
    for (int r = 0; r < 8; ++r) {
      const int row = r + 8 * khi;
      sHid[wv][row][nt * 16 + mlo]      = f2bf(fmaxf(acc0[nt][r] + bias, 0.0f));
      sHid[wv][16 + row][nt * 16 + mlo] = f2bf(fmaxf(acc1[nt][r] + bias, 0.0f));
    }
  }
  __syncthreads();

  // ---- layer 2: (32x128) @ (128x128), K-steps = 4 ------------------------
  const bf16_t* hrow0 = &sHid[wv][mlo][0];
  const bf16_t* hrow1 = &sHid[wv][16 + mlo][0];
  const v16bf* W2 = (const v16bf*)We2f;
  v8f acc20[8] = {}, acc21[8] = {};
  for (int ks = 0; ks < 4; ++ks) {
    v16bf a0 = load_afrag(hrow0, ks, lane);
    v16bf a1 = load_afrag(hrow1, ks, lane);
    v16bf b[8];
#pragma unroll
    for (int nt = 0; nt < 8; ++nt) b[nt] = W2[(ks * 8 + nt) * 32 + lane];
#pragma unroll
    for (int nt = 0; nt < 8; ++nt) {
      acc20[nt] = WMMA_BF16(a0, b[nt], acc20[nt]);
      acc21[nt] = WMMA_BF16(a1, b[nt], acc21[nt]);
    }
  }
  // ---- bias + relu + atomic scatter-add to agg[dest] ---------------------
#pragma unroll
  for (int nt = 0; nt < 8; ++nt) {
    const float bias = be2[nt * 16 + mlo];
    const int col = nt * 16 + mlo;
#pragma unroll
    for (int r = 0; r < 8; ++r) {
      const int row = r + 8 * khi;
      float v0 = fmaxf(acc20[nt][r] + bias, 0.0f);
      float v1 = fmaxf(acc21[nt][r] + bias, 0.0f);
      atomicAdd(agg + (size_t)sDst[wv][row] * D_HID + col, v0);
      atomicAdd(agg + (size_t)sDst[wv][16 + row] * D_HID + col, v1);
    }
  }
}

// ---------------------------------------------------------------------------
// Node block: per wave, a 16-node tile.  cat(x, agg) -> gate & update GEMMs.
// Writes x_weights to d_out, x_new scattered into pooled[batch].
// ---------------------------------------------------------------------------
__global__ __launch_bounds__(128) void in455_node(
    const float* __restrict__ x, const float* __restrict__ agg,
    const int* __restrict__ batch,
    const bf16_t* __restrict__ Wgf, const float* __restrict__ bg,
    const bf16_t* __restrict__ Wnf, const float* __restrict__ bn,
    float* __restrict__ xw_out, float* __restrict__ pooled,
    float* __restrict__ counts) {
  __shared__ __align__(16) bf16_t sCat[4][16][ROW3];
  __shared__ int sBat[4][16];

  const int lane = threadIdx.x & 31;
  const int wv   = threadIdx.x >> 5;
  const int mlo  = lane & 15;
  const int khi  = lane >> 4;
  const int tile0 = (blockIdx.x * 4 + wv) * 16;
  const bool act = (tile0 < N_NODES);      // full tiles only (50000 = 3125*16)
  const int tile = act ? tile0 : 0;

  if (lane < 16) {
    int b = batch[tile + lane];
    sBat[wv][lane] = b;
    if (act) atomicAdd(counts + b, 1.0f);  // each node counted once
  }
#pragma unroll 4
  for (int m = 0; m < 16; ++m) {
    const int n = tile + m;
    float4 vx = ((const float4*)(x   + (size_t)n * D_IN))[lane];
    float4 va = ((const float4*)(agg + (size_t)n * D_HID))[lane];
    *(unsigned long long*)&sCat[wv][m][lane * 4]        = pack4(vx);
    *(unsigned long long*)&sCat[wv][m][D_IN + lane * 4] = pack4(va);
  }
  __syncthreads();

  const bf16_t* arow = &sCat[wv][mlo][0];
  const v16bf* WG = (const v16bf*)Wgf;
  const v16bf* WN = (const v16bf*)Wnf;
  v8f g[8] = {}, nv[8] = {};
  for (int ks = 0; ks < 8; ++ks) {   // K = 256
    v16bf a = load_afrag(arow, ks, lane);
#pragma unroll
    for (int nt = 0; nt < 8; ++nt) {
      v16bf bG = WG[(ks * 8 + nt) * 32 + lane];
      v16bf bN = WN[(ks * 8 + nt) * 32 + lane];
      g[nt]  = WMMA_BF16(a, bG, g[nt]);
      nv[nt] = WMMA_BF16(a, bN, nv[nt]);
    }
  }

#pragma unroll
  for (int nt = 0; nt < 8; ++nt) {
    const int col = nt * 16 + mlo;
    const float bgv = bg[col], bnv = bn[col];
#pragma unroll
    for (int r = 0; r < 8; ++r) {
      const int row = r + 8 * khi;
      const int n = tile + row;
      float gate = 1.0f / (1.0f + __expf(-(g[nt][r] + bgv)));
      float xnew = gate * fmaxf(nv[nt][r] + bnv, 0.0f);
      if (act) {
        xw_out[(size_t)n * D_HID + col] = gate;                   // x_weights
        atomicAdd(pooled + (size_t)sBat[wv][row] * D_HID + col, xnew);
      }
    }
  }
}

// ---------------------------------------------------------------------------
// Global block: u_new = relu(cat(u, pooled/count) @ Wu + bu).  Tiny -> VALU.
// ---------------------------------------------------------------------------
__global__ __launch_bounds__(64) void in455_global(
    const float* __restrict__ u, const float* __restrict__ pooled,
    const float* __restrict__ counts, const float* __restrict__ Wu,
    const float* __restrict__ bu, float* __restrict__ out) {
  const int gph = blockIdx.x, t = threadIdx.x;
  float acc = bu[t];
  const float* ug = u + (size_t)gph * D_GLOB;
#pragma unroll 8
  for (int j = 0; j < D_GLOB; ++j) acc += ug[j] * Wu[(size_t)j * D_GLOB + t];
  const float inv = 1.0f / fmaxf(counts[gph], 1.0f);
  const float* pg = pooled + (size_t)gph * D_HID;
#pragma unroll 8
  for (int j = 0; j < D_HID; ++j)
    acc += (pg[j] * inv) * Wu[(size_t)(D_GLOB + j) * D_GLOB + t];
  out[(size_t)gph * D_GLOB + t] = fmaxf(acc, 0.0f);
}

// ---------------------------------------------------------------------------
extern "C" void kernel_launch(void* const* d_in, const int* in_sizes, int n_in,
                              void* d_out, int out_size, void* d_ws,
                              size_t ws_size, hipStream_t stream) {
  const float* x     = (const float*)d_in[0];
  const int*   eidx  = (const int*)d_in[1];
  const float* eattr = (const float*)d_in[2];
  const float* u     = (const float*)d_in[3];
  const int*   batch = (const int*)d_in[4];
  const float* We1   = (const float*)d_in[5];
  const float* be1   = (const float*)d_in[6];
  const float* We2   = (const float*)d_in[7];
  const float* be2   = (const float*)d_in[8];
  const float* Wg    = (const float*)d_in[9];
  const float* bg    = (const float*)d_in[10];
  const float* Wn    = (const float*)d_in[11];
  const float* bn    = (const float*)d_in[12];
  const float* Wu    = (const float*)d_in[13];
  const float* bu    = (const float*)d_in[14];
  float* out = (float*)d_out;

  // workspace layout (bytes): agg | pooled | counts | bf16 weight fragments
  float* agg    = (float*)d_ws;                              // 50000*128
  float* pooled = agg + (size_t)N_NODES * D_HID;             // 512*128
  float* counts = pooled + (size_t)N_GRAPHS * D_HID;         // 512
  bf16_t* We1f  = (bf16_t*)(counts + N_GRAPHS);              // 9*8*512 halves
  bf16_t* We2f  = We1f + 9 * 8 * 512;                        // 4*8*512
  bf16_t* Wgf   = We2f + 4 * 8 * 512;                        // 8*8*512
  bf16_t* Wnf   = Wgf + 8 * 8 * 512;                         // 8*8*512

  const size_t nz = (size_t)N_NODES * D_HID + (size_t)N_GRAPHS * D_HID + N_GRAPHS;
  in455_zero<<<2048, 256, 0, stream>>>(agg, nz);

  in455_prep_w<<<(9 * 8 * 32 + 255) / 256, 256, 0, stream>>>(We1, We1f, 9);
  in455_prep_w<<<(4 * 8 * 32 + 255) / 256, 256, 0, stream>>>(We2, We2f, 4);
  in455_prep_w<<<(8 * 8 * 32 + 255) / 256, 256, 0, stream>>>(Wg, Wgf, 8);
  in455_prep_w<<<(8 * 8 * 32 + 255) / 256, 256, 0, stream>>>(Wn, Wnf, 8);

  // 64 edges per block (2 waves x 32 edges)
  in455_edge<<<N_EDGES / 64, 64, 0, stream>>>(x, eidx, eattr, We1f, be1,
                                              We2f, be2, agg);

  const int node_blocks = (N_NODES / 16 + 3) / 4;  // 782
  in455_node<<<node_blocks, 128, 0, stream>>>(
      x, agg, batch, Wgf, bg, Wnf, bn,
      out + (size_t)N_GRAPHS * D_GLOB,  // x_weights after u_new
      pooled, counts);

  in455_global<<<N_GRAPHS, 64, 0, stream>>>(u, pooled, counts, Wu, bu, out);
}
